// AdaptiveSystem_20770461844031
// MI455X (gfx1250) — matrix operand order repaired and checked by reading, not compiled, not run on hardware
//
#include <hip/hip_runtime.h>
#include <hip/hip_bf16.h>

// ---------------------------------------------------------------------------
// MoE-of-2 CNN experts on MI455X (gfx1250), WMMA implicit-GEMM formulation.
//
// conv1: [64,3,224,224] -> [64,112,112,64]   GEMM M=802816 N=64  K=27(->32)
// conv2: [64,112,112,64]-> GAP [64,128]      GEMM M=200704 N=128 K=576
// finalize: GAP-mean, FC 128->2, softmax gate, blend, freq.
//
// ~65 GFLOP total, ~0.9 GB HBM traffic -> WMMA f16 (f32 accum) is the right
// precision/path; intermediate stored NHWC-f16 so conv2 A-fragments are two
// contiguous b128 loads per lane. Conv2 weights (144 KB repacked) staged
// through LDS in three 48 KB groups. Store addressing fully strength-reduced
// (tile-invariant (b,oy) hoisted out of the per-element path).
// ---------------------------------------------------------------------------

typedef _Float16 half_t;
typedef __attribute__((ext_vector_type(16))) _Float16 v16h;
typedef __attribute__((ext_vector_type(8)))  _Float16 v8h;
typedef __attribute__((ext_vector_type(8)))  float    v8f;

#define H1_HALFS   (64u * 112u * 112u * 64u)          // 51,380,224 halves
#define BP2_HALFS  (18 * 8 * 32 * 16)                 // 73,728 halves (144 KB)
#define BP1_HALFS  (4 * 32 * 16)                      // 2,048 halves  (4 KB)
#define GRP_HALFS  (6 * 8 * 32 * 16)                  // 24,576 halves (48 KB LDS)

// -------------------------- weight repacking -------------------------------
// B-fragment layout (16-bit B 32x16): lanes 0-15 hold column N=lane, K=0..7,16..23
// in halves 0..15; lanes 16-31 hold N=lane-16, K=8..15,24..31.
// conv1 K-order: k = (ky*3+kx)*3 + c   (K=27, padded to 32 with zeros)
// conv2 K-order: k = (ky*3+kx)*64 + ic (K=576 exactly = 18*32)

__global__ __launch_bounds__(256) void repack1_kernel(const float* __restrict__ w1,
                                                      half_t* __restrict__ bp) {
  int idx = blockIdx.x * 256 + threadIdx.x;            // [0, 4*32*16)
  if (idx >= BP1_HALFS) return;
  int h    = idx & 15;
  int lane = (idx >> 4) & 31;
  int nt   = idx >> 9;
  int k    = ((h >= 8) ? 16 : 0) + ((lane >= 16) ? 8 : 0) + (h & 7);
  int n    = nt * 16 + (lane & 15);
  half_t v = (half_t)0.0f;
  if (k < 27) {
    int tap = k / 3, c = k % 3;
    v = (half_t)w1[((n * 3 + c) * 3 + tap / 3) * 3 + (tap % 3)];  // OIHW
  }
  bp[idx] = v;
}

__global__ __launch_bounds__(256) void repack2_kernel(const float* __restrict__ w2,
                                                      half_t* __restrict__ bp) {
  int idx = blockIdx.x * 256 + threadIdx.x;            // [0, 18*8*32*16)
  if (idx >= BP2_HALFS) return;
  int h    = idx & 15;
  int lane = (idx >> 4) & 31;
  int nt   = (idx >> 9) & 7;
  int kt   = idx >> 12;
  int k    = kt * 32 + ((h >= 8) ? 16 : 0) + ((lane >= 16) ? 8 : 0) + (h & 7);
  int n    = nt * 16 + (lane & 15);
  int ic   = k & 63, tap = k >> 6;
  bp[idx] = (half_t)w2[((n * 64 + ic) * 3 + tap / 3) * 3 + (tap % 3)];  // OIHW
}

__global__ __launch_bounds__(256) void zero_kernel(float* __restrict__ p, int n) {
  int i = blockIdx.x * 256 + threadIdx.x;
  if (i < n) p[i] = 0.0f;
}

// ------------------------------- conv1 -------------------------------------
// One wave = one 16-row M tile. 112 % 16 == 0, so all 16 rows of a tile share
// (b, oy) and differ only in ox -> per-element store addressing is additive.
__global__ __launch_bounds__(256) void conv1_kernel(const float* __restrict__ x,
                                                    const half_t* __restrict__ bpack,
                                                    const float* __restrict__ bias,
                                                    half_t* __restrict__ h1) {
  const int lane  = threadIdx.x & 31;
  const int wave  = threadIdx.x >> 5;
  const int m0    = (blockIdx.x * 8 + wave) * 16;
  const int lane15 = lane & 15;
  const int hi8    = (lane >= 16) ? 8 : 0;

  // Tile-invariant decomposition (one division chain per wave, not per store).
  const int b0  = m0 / (112 * 112);
  const int sp0 = m0 % (112 * 112);
  const int oy  = sp0 / 112;
  const int ox0 = sp0 % 112;

  // A fragment: 16-bit A 16x32 layout; lane row ox = ox0 + lane15 at (b0, oy).
  const int ox = ox0 + lane15;
  const int kadd = hi8;                               // K base offset per half-wave
  v16h a;
#pragma unroll
  for (int h = 0; h < 16; ++h) {
    int k = h + ((h >= 8) ? 8 : 0) + kadd;
    half_t v = (half_t)0.0f;
    if (k < 27) {
      int tap = k / 3, c = k % 3;
      int iy = 2 * oy + tap / 3, ix = 2 * ox + tap % 3;  // SAME: pad_lo=0
      if (iy < 224 && ix < 224)
        v = (half_t)x[((b0 * 3 + c) * 224 + iy) * 224 + ix];
    }
    a[h] = v;
  }

  // NHWC-f16 output base for this tile's row (b0, oy, ox0..ox0+15).
  half_t* const obase = h1 + ((size_t)(b0 * 112 + oy) * 112 + ox0) * 64;

#pragma unroll
  for (int nt = 0; nt < 4; ++nt) {
    v16h bm = *(const v16h*)(bpack + (nt * 32 + lane) * 16);
    v8f acc = {};
    acc = __builtin_amdgcn_wmma_f32_16x16x32_f16(false, a, false, bm,
                                                 (short)0, acc, false, false);
    const int oc = nt * 16 + lane15;
    const float bs = bias[oc];
    half_t* op = obase + (size_t)hi8 * 64 + oc;       // D: M = r + 8*(lane>=16)
#pragma unroll
    for (int r = 0; r < 8; ++r) {
      float val = acc[r] + bs;
      val = val > 0.0f ? val : 0.0f;                  // ReLU
      op[(size_t)r * 64] = (half_t)val;               // additive addressing
    }
  }
}

// ------------------------------- conv2 -------------------------------------
// One wave = one 16-row M tile (3136%16==0 -> tile stays in one image), all
// 8 N-tiles (128 channels). K loop = 18 steps of 32, weights staged via LDS
// in three 48 KB groups. Epilogue fuses bias+ReLU+GAP (shfl fold + atomics).
__global__ __launch_bounds__(256) void conv2_kernel(const half_t* __restrict__ h1,
                                                    const half_t* __restrict__ bpack2,
                                                    const float* __restrict__ bias,
                                                    float* __restrict__ gap) {
  __shared__ half_t lds[GRP_HALFS];                   // 48 KB
  const int lane = threadIdx.x & 31;
  const int wave = threadIdx.x >> 5;
  const int m0   = (blockIdx.x * 8 + wave) * 16;
  const int lane15 = lane & 15;

  // Tile-invariant image index (3136 % 16 == 0).
  const int b0  = m0 / (56 * 56);
  const int sp0 = m0 % (56 * 56);
  // Per-lane spatial position (tile may straddle two oy rows: generic decode).
  const int sp  = sp0 + lane15;
  const int oy  = sp / 56, ox = sp % 56;
  const int icsel = (lane >= 16) ? 8 : 0;

  v8f acc[8];
#pragma unroll
  for (int nt = 0; nt < 8; ++nt) acc[nt] = (v8f){};

  for (int g = 0; g < 3; ++g) {
    __syncthreads();                                  // previous group consumed
    const half_t* src = bpack2 + g * GRP_HALFS;
    for (int i = threadIdx.x; i < GRP_HALFS / 8; i += 256)
      *(v8h*)(lds + i * 8) = *(const v8h*)(src + i * 8);
    __syncthreads();

#pragma unroll
    for (int kk = 0; kk < 6; ++kk) {
      const int kt = g * 6 + kk;                      // K block = [kt*32, kt*32+32)
      const int tap = kt >> 1;
      const int iy = 2 * oy + tap / 3, ix = 2 * ox + tap % 3;
      v16h a;
      if (iy < 112 && ix < 112) {
        // NHWC-f16: two contiguous 16B runs cover this lane's 16 K values.
        const half_t* ap = h1 + (((size_t)(b0 * 112 + iy) * 112 + ix) * 64
                                 + (kt & 1) * 32 + icsel);
        v8h r0 = *(const v8h*)(ap);
        v8h r1 = *(const v8h*)(ap + 16);
#pragma unroll
        for (int h = 0; h < 8; ++h) { a[h] = r0[h]; a[h + 8] = r1[h]; }
      } else {
        a = (v16h){};                                 // SAME high-edge padding
      }
#pragma unroll
      for (int nt = 0; nt < 8; ++nt) {
        v16h bm = *(const v16h*)(lds + ((kk * 8 + nt) * 32 + lane) * 16);
        acc[nt] = __builtin_amdgcn_wmma_f32_16x16x32_f16(false, a, false, bm,
                                                         (short)0, acc[nt],
                                                         false, false);
      }
    }
  }

  // bias + ReLU + partial GAP: fold 8 rows, then lane<->lane+16 (same column).
#pragma unroll
  for (int nt = 0; nt < 8; ++nt) {
    const int oc = nt * 16 + lane15;
    const float bs = bias[oc];
    float p = 0.0f;
#pragma unroll
    for (int r = 0; r < 8; ++r) {
      float v = acc[nt][r] + bs;
      p += (v > 0.0f ? v : 0.0f);
    }
    p += __shfl_xor(p, 16, 32);
    if (lane < 16) atomicAdd(&gap[b0 * 128 + oc], p);
  }
}

// ------------------------------ finalize -----------------------------------
__global__ __launch_bounds__(64) void finalize_kernel(const float* __restrict__ gap_t,
                                                      const float* __restrict__ gap_f,
                                                      const float* __restrict__ t_wf,
                                                      const float* __restrict__ t_bf,
                                                      const float* __restrict__ f_wf,
                                                      const float* __restrict__ f_bf,
                                                      float* __restrict__ out) {
  __shared__ float use_s[64];
  const int b = threadIdx.x;                          // blockDim.x == 64
  float t0 = t_bf[0], t1 = t_bf[1], f0 = f_bf[0], f1 = f_bf[1];
  const float inv = 1.0f / 3136.0f;                   // GAP mean over 56*56
  for (int c = 0; c < 128; ++c) {
    float gt = gap_t[b * 128 + c] * inv;
    float gf = gap_f[b * 128 + c] * inv;
    t0 += gt * t_wf[c * 2 + 0];
    t1 += gt * t_wf[c * 2 + 1];
    f0 += gf * f_wf[c * 2 + 0];
    f1 += gf * f_wf[c * 2 + 1];
  }
  float mx = fmaxf(t0, t1);
  float e0 = __expf(t0 - mx), e1 = __expf(t1 - mx);
  float conf = fmaxf(e0, e1) / (e0 + e1);
  bool use2 = conf <= 0.9f;
  out[b * 2 + 0] = use2 ? 0.7f * t0 + 0.3f * f0 : t0;
  out[b * 2 + 1] = use2 ? 0.7f * t1 + 0.3f * f1 : t1;
  use_s[b] = use2 ? 1.0f : 0.0f;
  __syncthreads();
  if (b == 0) {
    float s = 0.0f;
    for (int i = 0; i < 64; ++i) s += use_s[i];
    out[128] = s * (1.0f / 64.0f);
  }
}

// ------------------------------ launcher -----------------------------------
extern "C" void kernel_launch(void* const* d_in, const int* in_sizes, int n_in,
                              void* d_out, int out_size, void* d_ws, size_t ws_size,
                              hipStream_t stream) {
  const float* x    = (const float*)d_in[0];
  const float* t_w1 = (const float*)d_in[1];
  const float* t_b1 = (const float*)d_in[2];
  const float* t_w2 = (const float*)d_in[3];
  const float* t_b2 = (const float*)d_in[4];
  const float* t_wf = (const float*)d_in[5];
  const float* t_bf = (const float*)d_in[6];
  const float* f_w1 = (const float*)d_in[7];
  const float* f_b1 = (const float*)d_in[8];
  const float* f_w2 = (const float*)d_in[9];
  const float* f_b2 = (const float*)d_in[10];
  const float* f_wf = (const float*)d_in[11];
  const float* f_bf = (const float*)d_in[12];
  float* out = (float*)d_out;

  // Workspace layout (~103 MB): h1 reused by both experts (stream serializes).
  char* ws = (char*)d_ws;
  size_t off = 0;
  half_t* h1     = (half_t*)(ws + off); off += (size_t)H1_HALFS * 2;   // 102,760,448
  half_t* bpack2 = (half_t*)(ws + off); off += (size_t)BP2_HALFS * 2;  // 147,456
  half_t* bpack1 = (half_t*)(ws + off); off += 4096;                   // 4,096 (padded)
  float*  gap_t  = (float*)(ws + off);  off += 64 * 128 * 4;
  float*  gap_f  = (float*)(ws + off);  off += 64 * 128 * 4;
  (void)ws_size; (void)in_sizes; (void)n_in; (void)out_size;

  const int C1_BLOCKS = (64 * 112 * 112) / 16 / 8;   // 6272
  const int C2_BLOCKS = (64 * 56 * 56) / 16 / 8;     // 1568

  // gap_t and gap_f are contiguous: zero both in one launch.
  zero_kernel<<<64, 256, 0, stream>>>(gap_t, 2 * 64 * 128);

  // ---- texture expert ----
  repack1_kernel<<<8, 256, 0, stream>>>(t_w1, bpack1);
  conv1_kernel<<<C1_BLOCKS, 256, 0, stream>>>(x, bpack1, t_b1, h1);
  repack2_kernel<<<288, 256, 0, stream>>>(t_w2, bpack2);
  conv2_kernel<<<C2_BLOCKS, 256, 0, stream>>>(h1, bpack2, t_b2, gap_t);

  // ---- frequency expert (reuses h1/bpack buffers) ----
  repack1_kernel<<<8, 256, 0, stream>>>(f_w1, bpack1);
  conv1_kernel<<<C1_BLOCKS, 256, 0, stream>>>(x, bpack1, f_b1, h1);
  repack2_kernel<<<288, 256, 0, stream>>>(f_w2, bpack2);
  conv2_kernel<<<C2_BLOCKS, 256, 0, stream>>>(h1, bpack2, f_b2, gap_f);

  // ---- gate + blend + freq ----
  finalize_kernel<<<1, 64, 0, stream>>>(gap_t, gap_f, t_wf, t_bf, f_wf, f_bf, out);
}
